// CrossAttentionFusion_36189394436148
// MI455X (gfx1250) — matrix operand order repaired and checked by reading, not compiled, not run on hardware
//
#include <hip/hip_runtime.h>

// ---------------------------------------------------------------------------
// CrossAttentionFusion for MI455X (gfx1250, wave32, WMMA).
// seq_len==1 attention => softmax prob == 1, so
//   MHA(xq, xkv) == (xkv @ Wv^T + bv) @ Wo^T + bo   (Q/K projections dead).
// Whole model = chain of GEMMs + bias/relu + residual-LayerNorm.
// GEMM: v_wmma_f32_16x16x32_bf16, fp32 accum; fp32->bf16 via native packed
// converts (__builtin_convertvector, RNE) while staging into double-buffered
// LDS; one barrier per K-step; k+2 tile prefetched (global_prefetch_b8).
// ---------------------------------------------------------------------------

typedef __bf16 bf16;
typedef __attribute__((ext_vector_type(16))) __bf16 v16bf;
typedef __attribute__((ext_vector_type(8)))  __bf16 v8bf;
typedef __attribute__((ext_vector_type(4)))  __bf16 v4bf;
typedef __attribute__((ext_vector_type(8)))  float  v8f;
typedef __attribute__((ext_vector_type(4)))  float  v4f;

static constexpr int B_   = 4096;
static constexpr int D_   = 1024;
static constexpr int DFF_ = 4096;
static constexpr int L_   = 2;

// ---------------------------------------------------------------------------
// GEMM: C[M,N] = act(A[M,K] @ W[N,K]^T + bias[N]); A,W fp32 (K-contiguous).
// Block tile 128x128, K-step 32, double-buffered LDS.
// 8 waves arranged 2(M) x 4(N); each wave: 64x32 tile = 4x2 WMMA fragments
// => 8 v_wmma per K-step from 6 fragment fetches.
// ---------------------------------------------------------------------------
#define TBM 128
#define TBN 128
#define TBK 32
#define LPAD 8
#define LSTR (TBK + LPAD)   // 40 bf16 = 80B row stride (16B aligned, conflict-free)

template <int RELU>
__global__ __launch_bounds__(256)
void gemm_bias_act(const float* __restrict__ A, const float* __restrict__ W,
                   const float* __restrict__ bias, float* __restrict__ C,
                   int M, int N, int K)
{
  __shared__ bf16 sA[2][TBM * LSTR];   // 2 x 10KB
  __shared__ bf16 sB[2][TBN * LSTR];   // 2 x 10KB

  const int tid  = threadIdx.x;
  const int lane = tid & 31;
  const int wid  = tid >> 5;       // 0..7
  const int wm   = wid & 1;        // 2 waves along M (64 rows each)
  const int wn   = wid >> 1;       // 4 waves along N (32 cols each)
  const int m0   = blockIdx.y * TBM;
  const int n0   = blockIdx.x * TBN;
  const int kh   = lane >> 4;      // lane half (0/1)
  const int l16  = lane & 15;

  const int q     = tid & 7;       // float4 slot within a 32-wide K row
  const int rbase = tid >> 3;      // 0..31 (row pass base)

  v8f acc[4][2] = {};
  v4f regA[4], regB[4];

  const float* aBase = A + (size_t)(m0 + rbase) * K + q * 4;
  const float* wBase = W + (size_t)(n0 + rbase) * K + q * 4;
  const size_t rowStep = (size_t)32 * K;

  // prologue: stage K-tile 0 into registers
#pragma unroll
  for (int r = 0; r < 4; ++r) {
    regA[r] = *(const v4f*)(aBase + r * rowStep);
    regB[r] = *(const v4f*)(wBase + r * rowStep);
  }

  int p = 0;
  for (int kb = 0; kb < K; kb += TBK) {
    // ---- commit staged registers to LDS buffer p (packed fp32 -> bf16) ----
#pragma unroll
    for (int r = 0; r < 4; ++r) {
      *(v4bf*)&sA[p][(rbase + r * 32) * LSTR + q * 4] =
          __builtin_convertvector(regA[r], v4bf);
      *(v4bf*)&sB[p][(rbase + r * 32) * LSTR + q * 4] =
          __builtin_convertvector(regB[r], v4bf);
    }
    __syncthreads();

    // ---- issue next tile's global loads early (overlap with WMMA) ----
    if (kb + TBK < K) {
      const float* aN = aBase + kb + TBK;
      const float* wN = wBase + kb + TBK;
#pragma unroll
      for (int r = 0; r < 4; ++r) {
        regA[r] = *(const v4f*)(aN + r * rowStep);
        regB[r] = *(const v4f*)(wN + r * rowStep);
      }
      if (kb + 2 * TBK < K) {  // prewarm L2 for tile k+2 (global_prefetch_b8)
        __builtin_prefetch(aBase + kb + 2 * TBK, 0, 1);
        __builtin_prefetch(wBase + kb + 2 * TBK, 0, 1);
      }
    }

    // ---- fragments per ISA 7.12.2 lane layouts ----
    v16bf afrag[4], bfrag[2];
#pragma unroll
    for (int im = 0; im < 4; ++im) {
      const int row = wm * 64 + im * 16 + l16;
      // lanes 0-15: elems 0..7 = K0..7,  8..15 = K16..23
      // lanes16-31: elems 0..7 = K8..15, 8..15 = K24..31
      const v8bf lo = *(const v8bf*)&sA[p][row * LSTR + kh * 8];
      const v8bf hi = *(const v8bf*)&sA[p][row * LSTR + 16 + kh * 8];
      afrag[im] = __builtin_shufflevector(lo, hi, 0,1,2,3,4,5,6,7,8,9,10,11,12,13,14,15);
    }
#pragma unroll
    for (int in = 0; in < 2; ++in) {
      const int row = wn * 32 + in * 16 + l16;   // N row of W (K-contiguous)
      // lanes 0-15: elems 0..7 = K0..7,  8..15 = K16..23; lanes 16-31: K8..15, K24..31
      const v8bf lo = *(const v8bf*)&sB[p][row * LSTR + kh * 8];
      const v8bf hi = *(const v8bf*)&sB[p][row * LSTR + 16 + kh * 8];
      bfrag[in] = __builtin_shufflevector(lo, hi, 0,1,2,3,4,5,6,7,8,9,10,11,12,13,14,15);
    }

#pragma unroll
    for (int im = 0; im < 4; ++im)
#pragma unroll
      for (int in = 0; in < 2; ++in)
        acc[im][in] = __builtin_amdgcn_wmma_f32_16x16x32_bf16(
            false, afrag[im], false, bfrag[in], (short)0, acc[im][in], false, false);

    p ^= 1;
  }

  // ---- epilogue: C/D layout -> VGPR r holds M = r + 8*kh, N = l16 ----
#pragma unroll
  for (int in = 0; in < 2; ++in) {
    const int n = n0 + wn * 32 + in * 16 + l16;
    const float bv = bias[n];
#pragma unroll
    for (int im = 0; im < 4; ++im) {
      const int mbase = m0 + wm * 64 + im * 16 + kh * 8;
#pragma unroll
      for (int r = 0; r < 8; ++r) {
        float val = acc[im][in][r] + bv;
        if (RELU) val = fmaxf(val, 0.0f);
        C[(size_t)(mbase + r) * N + n] = val;
      }
    }
  }
}

// ---------------------------------------------------------------------------
// out[row, outOffset + :D] = LayerNorm(x[row] + res[row]) * g + b
// One 256-thread block per row, D = 1024 fixed (4 elems/thread).
// Safe in-place (all reads complete before the post-reduction writes).
// ---------------------------------------------------------------------------
__global__ __launch_bounds__(256)
void ln_residual(const float* __restrict__ x, const float* __restrict__ res,
                 const float* __restrict__ g, const float* __restrict__ b,
                 float* __restrict__ out, int outOffset, int outStride)
{
  __shared__ float s1[256];
  __shared__ float s2[256];
  const int row = blockIdx.x;
  const int tid = threadIdx.x;

  const float4 xv = *(const float4*)(x   + (size_t)row * D_ + tid * 4);
  const float4 rv = *(const float4*)(res + (size_t)row * D_ + tid * 4);
  float vals[4] = {xv.x + rv.x, xv.y + rv.y, xv.z + rv.z, xv.w + rv.w};

  float s  = vals[0] + vals[1] + vals[2] + vals[3];
  float sq = vals[0]*vals[0] + vals[1]*vals[1] + vals[2]*vals[2] + vals[3]*vals[3];
  s1[tid] = s; s2[tid] = sq;
  __syncthreads();
  for (int off = 128; off > 0; off >>= 1) {
    if (tid < off) { s1[tid] += s1[tid + off]; s2[tid] += s2[tid + off]; }
    __syncthreads();
  }
  const float mu   = s1[0] * (1.0f / D_);
  const float var  = s2[0] * (1.0f / D_) - mu * mu;
  const float rstd = rsqrtf(var + 1e-5f);

  const float4 gv = *(const float4*)(g + tid * 4);
  const float4 bv = *(const float4*)(b + tid * 4);
  float* o = out + (size_t)row * outStride + outOffset + tid * 4;
  o[0] = (vals[0] - mu) * rstd * gv.x + bv.x;
  o[1] = (vals[1] - mu) * rstd * gv.y + bv.y;
  o[2] = (vals[2] - mu) * rstd * gv.z + bv.z;
  o[3] = (vals[3] - mu) * rstd * gv.w + bv.w;
}

// ---------------------------------------------------------------------------
extern "C" void kernel_launch(void* const* d_in, const int* in_sizes, int n_in,
                              void* d_out, int out_size, void* d_ws, size_t ws_size,
                              hipStream_t stream)
{
  (void)in_sizes; (void)n_in; (void)out_size; (void)ws_size;

  const float* vision  = (const float*)d_in[0];
  const float* text    = (const float*)d_in[1];
  const float* vw      = (const float*)d_in[2];
  const float* vb      = (const float*)d_in[3];
  const float* tw      = (const float*)d_in[4];
  const float* tb      = (const float*)d_in[5];
  const float* sa_wqkv = (const float*)d_in[6];
  const float* sa_bqkv = (const float*)d_in[7];
  const float* sa_wo   = (const float*)d_in[8];
  const float* sa_bo   = (const float*)d_in[9];
  const float* ca_wqkv = (const float*)d_in[10];
  const float* ca_bqkv = (const float*)d_in[11];
  const float* ca_wo   = (const float*)d_in[12];
  const float* ca_bo   = (const float*)d_in[13];
  const float* ln1g    = (const float*)d_in[14];
  const float* ln1b    = (const float*)d_in[15];
  const float* ln2g    = (const float*)d_in[16];
  const float* ln2b    = (const float*)d_in[17];
  const float* ln3g    = (const float*)d_in[18];
  const float* ln3b    = (const float*)d_in[19];
  const float* fw1     = (const float*)d_in[20];
  const float* fb1     = (const float*)d_in[21];
  const float* fw2     = (const float*)d_in[22];
  const float* fb2     = (const float*)d_in[23];
  const float* fus_w1  = (const float*)d_in[24];
  const float* fus_b1  = (const float*)d_in[25];
  const float* fus_w2  = (const float*)d_in[26];
  const float* fus_b2  = (const float*)d_in[27];

  // Workspace layout (fp32): v, t, proj tmp, attn-v, attn-t, ffn hidden, fused.
  float* v     = (float*)d_ws;
  float* t     = v  + (size_t)B_ * D_;
  float* p     = t  + (size_t)B_ * D_;
  float* av    = p  + (size_t)B_ * D_;
  float* at    = av + (size_t)B_ * D_;
  float* h     = at + (size_t)B_ * D_;
  float* fused = h  + (size_t)B_ * DFF_;
  float* f1    = p;   // reuse proj tmp for fusion hidden

  dim3 blk(256, 1, 1);
  auto G = [&](const float* Ap, const float* Wp, const float* bp, float* Cp,
               int M, int N, int K, int relu) {
    dim3 grid(N / TBN, M / TBM, 1);
    if (relu) gemm_bias_act<1><<<grid, blk, 0, stream>>>(Ap, Wp, bp, Cp, M, N, K);
    else      gemm_bias_act<0><<<grid, blk, 0, stream>>>(Ap, Wp, bp, Cp, M, N, K);
  };
  auto LN = [&](const float* xp, const float* rp, const float* gp, const float* bp,
                float* op, int off, int stride) {
    ln_residual<<<dim3(B_, 1, 1), blk, 0, stream>>>(xp, rp, gp, bp, op, off, stride);
  };

  // Input projections
  G(vision, vw, vb, v, B_, D_, D_, 0);
  G(text,   tw, tb, t, B_, D_, D_, 0);

  for (int i = 0; i < L_; ++i) {
    const bool last = (i == L_ - 1);

    // ---- shared self-attention (seq1 => V-proj + O-proj only) ----
    const float* Wsav = sa_wqkv + (size_t)i * 3 * D_ * D_ + (size_t)2 * D_ * D_;
    const float* bsav = sa_bqkv + (size_t)i * 3 * D_ + 2 * D_;
    const float* Wso  = sa_wo + (size_t)i * D_ * D_;
    const float* bso  = sa_bo + (size_t)i * D_;
    G(v, Wsav, bsav, p,  B_, D_, D_, 0);
    G(p, Wso,  bso,  av, B_, D_, D_, 0);
    G(t, Wsav, bsav, p,  B_, D_, D_, 0);
    G(p, Wso,  bso,  at, B_, D_, D_, 0);
    LN(v, av, ln1g + i * D_, ln1b + i * D_, v, 0, D_);
    LN(t, at, ln1g + i * D_, ln1b + i * D_, t, 0, D_);

    // ---- shared cross-attention: output depends only on the KV side ----
    const float* Wcav = ca_wqkv + (size_t)i * 3 * D_ * D_ + (size_t)2 * D_ * D_;
    const float* bcav = ca_bqkv + (size_t)i * 3 * D_ + 2 * D_;
    const float* Wco  = ca_wo + (size_t)i * D_ * D_;
    const float* bco  = ca_bo + (size_t)i * D_;
    G(t, Wcav, bcav, p,  B_, D_, D_, 0);   // vc = O(V(t))
    G(p, Wco,  bco,  av, B_, D_, D_, 0);
    G(v, Wcav, bcav, p,  B_, D_, D_, 0);   // tc = O(V(v))
    G(p, Wco,  bco,  at, B_, D_, D_, 0);
    LN(v, av, ln2g + i * D_, ln2b + i * D_, v, 0, D_);
    LN(t, at, ln3g + i * D_, ln3b + i * D_, t, 0, D_);

    // ---- shared FFN ----
    const float* W1 = fw1 + (size_t)i * DFF_ * D_;
    const float* b1 = fb1 + (size_t)i * DFF_;
    const float* W2 = fw2 + (size_t)i * D_ * DFF_;
    const float* b2 = fb2 + (size_t)i * D_;
    G(v, W1, b1, h,  B_, DFF_, D_, 1);
    G(h, W2, b2, av, B_, D_, DFF_, 0);
    if (last) LN(v, av, ln2g + i * D_, ln2b + i * D_, fused, 0, 2 * D_);
    else      LN(v, av, ln2g + i * D_, ln2b + i * D_, v, 0, D_);
    G(t, W1, b1, h,  B_, DFF_, D_, 1);
    G(h, W2, b2, at, B_, D_, DFF_, 0);
    if (last) LN(t, at, ln3g + i * D_, ln3b + i * D_, fused, D_, 2 * D_);
    else      LN(t, at, ln3g + i * D_, ln3b + i * D_, t, 0, D_);
  }

  // ---- fusion head: relu(fused @ fus_w1^T + b) @ fus_w2^T + b ----
  G(fused, fus_w1, fus_b1, f1, B_, D_, 2 * D_, 1);
  G(f1,    fus_w2, fus_b2, (float*)d_out, B_, D_, D_, 0);
}
